// ChebNet_7876970020888
// MI455X (gfx1250) — matrix operand order repaired and checked by reading, not compiled, not run on hardware
//
#include <hip/hip_runtime.h>
#include <hip/hip_bf16.h>

typedef __attribute__((ext_vector_type(16))) _Float16 v16h;
typedef __attribute__((ext_vector_type(8)))  float    v8f;

#define NN  100000
#define NE  1600000
#define FIN 64
#define HID 16
#define CO  40

__global__ void k_zero(float* __restrict__ p, int n) {
  int i = blockIdx.x * blockDim.x + threadIdx.x;
  if (i < n) p[i] = 0.0f;
}

__global__ void k_deg(const float* __restrict__ ew, const int* __restrict__ dst,
                      float* __restrict__ deg) {
  int e = blockIdx.x * blockDim.x + threadIdx.x;
  if (e < NE) atomicAdd(&deg[dst[e]], ew[e]);
}

__global__ void k_dinv(float* __restrict__ deg) {
  int i = blockIdx.x * blockDim.x + threadIdx.x;
  if (i < NN) {
    float d = deg[i];
    deg[i] = (d > 0.0f) ? rsqrtf(fmaxf(d, 1e-30f)) : 0.0f;
  }
}

__global__ void k_norm(const float* __restrict__ ew, const int* __restrict__ src,
                       const int* __restrict__ dst, const float* __restrict__ dinv,
                       float* __restrict__ nrm) {
  int e = blockIdx.x * blockDim.x + threadIdx.x;
  if (e < NE) nrm[e] = -dinv[src[e]] * ew[e] * dinv[dst[e]];
}

// Scatter SpMM: Tx1[dst] += norm[e] * feat[src]. One thread = one edge x 4 features.
// feat rows are L2-resident (<=25.6MB), atomics land in L2.
template <int F, int LOG2C>
__global__ void k_spmm(const float* __restrict__ feat, const int* __restrict__ src,
                       const int* __restrict__ dst, const float* __restrict__ nrm,
                       float* __restrict__ out) {
  unsigned gid = blockIdx.x * blockDim.x + threadIdx.x;
  if (gid >= ((unsigned)NE << LOG2C)) return;
  unsigned e = gid >> LOG2C;
  unsigned c = (gid & ((1u << LOG2C) - 1u)) << 2;
  float w = nrm[e];
  const float4 v = *(const float4*)(feat + (size_t)src[e] * F + c);
  float* o = out + (size_t)dst[e] * F + c;
  atomicAdd(o + 0, w * v.x);
  atomicAdd(o + 1, w * v.y);
  atomicAdd(o + 2, w * v.z);
  atomicAdd(o + 3, w * v.w);
}

// Layer-1 dense: h = relu([x | Tx1](N x 128) @ W1flat(128 x 16) + b1)
// One wave per 16-row tile; 4 x v_wmma_f32_16x16x32_f16 over K=128.
__global__ void k_gemm1(const float* __restrict__ x, const float* __restrict__ tx1,
                        const float* __restrict__ W1, const float* __restrict__ b1,
                        float* __restrict__ h) {
  const int wave = (int)((blockIdx.x * blockDim.x + threadIdx.x) >> 5);
  if (wave >= NN / 16) return;
  const int lane = threadIdx.x & 31;
  const int m  = lane & 15;   // A row-in-tile / B,C,D column
  const int hi = lane >> 4;   // lane half selects K sub-ranges
  const size_t arow = (size_t)(wave * 16 + m);

  const float bias = b1[m];   // C/D column N == lane & 15
  v8f acc;
#pragma unroll
  for (int r = 0; r < 8; ++r) acc[r] = bias;

#pragma unroll
  for (int s = 0; s < 4; ++s) {
    const int kb = s * 32;
    // ISA 16-bit A 16x32 layout: lanes<16 hold K {0..7,16..23}, lanes>=16 {8..15,24..31}
    const int k0 = kb + hi * 8;
    const int k1 = kb + 16 + hi * 8;
    const float* pa0 = (k0 < FIN) ? (x + arow * FIN + k0) : (tx1 + arow * FIN + (k0 - FIN));
    const float* pa1 = (k1 < FIN) ? (x + arow * FIN + k1) : (tx1 + arow * FIN + (k1 - FIN));
    const float4 a0l = ((const float4*)pa0)[0], a0h = ((const float4*)pa0)[1];
    const float4 a1l = ((const float4*)pa1)[0], a1h = ((const float4*)pa1)[1];
    v16h A;
    A[0] = (_Float16)a0l.x;  A[1] = (_Float16)a0l.y;  A[2]  = (_Float16)a0l.z;  A[3]  = (_Float16)a0l.w;
    A[4] = (_Float16)a0h.x;  A[5] = (_Float16)a0h.y;  A[6]  = (_Float16)a0h.z;  A[7]  = (_Float16)a0h.w;
    A[8] = (_Float16)a1l.x;  A[9] = (_Float16)a1l.y;  A[10] = (_Float16)a1l.z;  A[11] = (_Float16)a1l.w;
    A[12] = (_Float16)a1h.x; A[13] = (_Float16)a1h.y; A[14] = (_Float16)a1h.z;  A[15] = (_Float16)a1h.w;

    // 16-bit B 32x16: lane = column; element j <-> klocal = 8*hi + j (+8 for j>=8), matching A
    v16h B;
    const int koff = kb + hi * 8;
#pragma unroll
    for (int j = 0; j < 16; ++j) B[j] = (_Float16)W1[(koff + j + (j >= 8 ? 8 : 0)) * HID + m];

    acc = __builtin_amdgcn_wmma_f32_16x16x32_f16(false, A, false, B, (short)0, acc,
                                                 false, false);
  }

#pragma unroll
  for (int r = 0; r < 8; ++r) {
    const int rowM = wave * 16 + r + hi * 8;  // C/D: lanes>=16 hold M = r+8
    const float v = acc[r];
    h[(size_t)rowM * HID + m] = (v > 0.0f) ? v : 0.0f;
  }
}

// Layer-2 dense: o = [h | Tx1h](N x 32) @ W2flat(32 x 40) + b2
// One wave per 16-row tile, 3 column tiles (16,16,8-masked), K=32 -> single WMMA each.
__global__ void k_gemm2(const float* __restrict__ h, const float* __restrict__ tx1h,
                        const float* __restrict__ W2, const float* __restrict__ b2,
                        float* __restrict__ o) {
  const int wave = (int)((blockIdx.x * blockDim.x + threadIdx.x) >> 5);
  if (wave >= NN / 16) return;
  const int lane = threadIdx.x & 31;
  const int m  = lane & 15;
  const int hi = lane >> 4;
  const size_t arow = (size_t)(wave * 16 + m);

  // A: K 0..15 from h, K 16..31 from Tx1h (each 8-run stays in one source)
  const float* pa0 = h    + arow * HID + hi * 8;
  const float* pa1 = tx1h + arow * HID + hi * 8;
  const float4 a0l = ((const float4*)pa0)[0], a0h = ((const float4*)pa0)[1];
  const float4 a1l = ((const float4*)pa1)[0], a1h = ((const float4*)pa1)[1];
  v16h A;
  A[0] = (_Float16)a0l.x;  A[1] = (_Float16)a0l.y;  A[2]  = (_Float16)a0l.z;  A[3]  = (_Float16)a0l.w;
  A[4] = (_Float16)a0h.x;  A[5] = (_Float16)a0h.y;  A[6]  = (_Float16)a0h.z;  A[7]  = (_Float16)a0h.w;
  A[8] = (_Float16)a1l.x;  A[9] = (_Float16)a1l.y;  A[10] = (_Float16)a1l.z;  A[11] = (_Float16)a1l.w;
  A[12] = (_Float16)a1h.x; A[13] = (_Float16)a1h.y; A[14] = (_Float16)a1h.z;  A[15] = (_Float16)a1h.w;

#pragma unroll
  for (int nt = 0; nt < 3; ++nt) {
    const int n = nt * 16 + m;
    const bool valid = (n < CO);
    v16h B;
#pragma unroll
    for (int j = 0; j < 16; ++j) {
      const int k = hi * 8 + j + (j >= 8 ? 8 : 0);
      B[j] = valid ? (_Float16)W2[k * CO + n] : (_Float16)0.0f;
    }
    const float bias = valid ? b2[n] : 0.0f;
    v8f acc;
#pragma unroll
    for (int r = 0; r < 8; ++r) acc[r] = bias;

    acc = __builtin_amdgcn_wmma_f32_16x16x32_f16(false, A, false, B, (short)0, acc,
                                                 false, false);
    if (valid) {
#pragma unroll
      for (int r = 0; r < 8; ++r)
        o[(size_t)(wave * 16 + r + hi * 8) * CO + n] = acc[r];
    }
  }
}

__global__ void k_lsm(const float* __restrict__ o, float* __restrict__ out) {
  int i = blockIdx.x * blockDim.x + threadIdx.x;
  if (i >= NN) return;
  const float* r = o + (size_t)i * CO;
  float mx = r[0];
#pragma unroll
  for (int j = 1; j < CO; ++j) mx = fmaxf(mx, r[j]);
  float s = 0.0f;
#pragma unroll
  for (int j = 0; j < CO; ++j) s += __expf(r[j] - mx);
  const float ls = __logf(s);
  float* w = out + (size_t)i * CO;
#pragma unroll
  for (int j = 0; j < CO; ++j) w[j] = r[j] - mx - ls;
}

extern "C" void kernel_launch(void* const* d_in, const int* in_sizes, int n_in,
                              void* d_out, int out_size, void* d_ws, size_t ws_size,
                              hipStream_t stream) {
  (void)in_sizes; (void)n_in; (void)out_size; (void)ws_size;
  const float* x  = (const float*)d_in[0];
  const float* ew = (const float*)d_in[1];
  const float* W1 = (const float*)d_in[2];   // (2,64,16) flat == 128x16 K-concat
  const float* b1 = (const float*)d_in[3];
  const float* W2 = (const float*)d_in[4];   // (2,16,40) flat == 32x40 K-concat
  const float* b2 = (const float*)d_in[5];
  const int* ei   = (const int*)d_in[6];
  const int* src = ei;
  const int* dst = ei + NE;

  float* ws   = (float*)d_ws;
  float* deg  = ws;                              // N       (becomes dinv in-place)
  float* nrm  = deg  + NN;                       // E
  float* tx1  = nrm  + NE;                       // N*64
  float* h    = tx1  + (size_t)NN * FIN;         // N*16
  float* tx1h = h    + (size_t)NN * HID;         // N*16
  float* o    = tx1h + (size_t)NN * HID;         // N*40   (total ~61.2 MB)

  const int B = 256;
  k_zero<<<(NN + B - 1) / B, B, 0, stream>>>(deg, NN);
  k_zero<<<(NN * FIN + B - 1) / B, B, 0, stream>>>(tx1, NN * FIN);
  k_zero<<<(NN * HID + B - 1) / B, B, 0, stream>>>(tx1h, NN * HID);

  k_deg <<<(NE + B - 1) / B, B, 0, stream>>>(ew, dst, deg);
  k_dinv<<<(NN + B - 1) / B, B, 0, stream>>>(deg);
  k_norm<<<(NE + B - 1) / B, B, 0, stream>>>(ew, src, dst, deg, nrm);

  k_spmm<64, 4><<<(NE * 16 + B - 1) / B, B, 0, stream>>>(x, src, dst, nrm, tx1);

  const int gemm_blocks = ((NN / 16) * 32 + B - 1) / B;  // one wave per 16-row tile
  k_gemm1<<<gemm_blocks, B, 0, stream>>>(x, tx1, W1, b1, h);

  k_spmm<16, 2><<<(NE * 4 + B - 1) / B, B, 0, stream>>>(h, src, dst, nrm, tx1h);

  k_gemm2<<<gemm_blocks, B, 0, stream>>>(h, tx1h, W2, b2, o);

  k_lsm<<<(NN + B - 1) / B, B, 0, stream>>>(o, (float*)d_out);
}
